// FastBloomAttention_87419764343112
// MI455X (gfx1250) — matrix operand order repaired and checked by reading, not compiled, not run on hardware
//
#include <hip/hip_runtime.h>

// ---------------- types ----------------
typedef __attribute__((ext_vector_type(16))) __bf16 v16bf;
typedef __attribute__((ext_vector_type(8)))  __bf16 v8bf;
typedef __attribute__((ext_vector_type(8)))  float  v8f;
typedef __attribute__((ext_vector_type(4)))  float  v4f;

// TDM descriptor groups (6-arg builtin signature: amdgpu-toolchain clang-23)
typedef __attribute__((ext_vector_type(4))) unsigned tdm_g0_t;
typedef __attribute__((ext_vector_type(8))) int      tdm_g1_t;
typedef __attribute__((ext_vector_type(4))) int      tdm_g2_t;
typedef __attribute__((ext_vector_type(8))) int      tdm_g4_t;

#define SEQ   2048
#define HDIM  128
#define NHEAD 16
#define HID   2048
#define NQKV  6144
#define NROWS 4096   // B*S

static __device__ __forceinline__ __bf16 f2bf(float f) {
    return (__bf16)f;          // native v_cvt on gfx1250
}

static __device__ __forceinline__ v16bf mk16(v8bf lo, v8bf hi) {
    v16bf r;
#pragma unroll
    for (int i = 0; i < 8; i++) { r[i] = lo[i]; r[i + 8] = hi[i]; }
    return r;
}

static __device__ __forceinline__ v8f wmma_bf16(v16bf a, v16bf b, v8f c) {
    return __builtin_amdgcn_wmma_f32_16x16x32_bf16(false, a, false, b, (short)0, c, false, false);
}

// ---------------------------------------------------------------------
// TDM: async 2D tile load global -> LDS with LDS row padding.
// D# built per CDNA5 ISA sec 8.3-8.6:
//   group0: count=1, lds_addr, 57b global_addr, type=2
//   group1: data_size=2B, pad_enable, pad_interval/amount,
//           tensor_dim0/1, tile_dim0/1, tensor_dim0_stride
// pad_interval code: 0=2,1=4,2=8,3=16,4=32,5=64 DWORDs between pads
// pad_amount  code: N-1 for N DWORDs of pad
// ---------------------------------------------------------------------
static __device__ __forceinline__ void tdm_load_2d(
    unsigned lds, const void* gptr,
    unsigned dim0, unsigned dim1,
    unsigned tile0, unsigned tile1,
    unsigned stride0,
    unsigned pad_interval, unsigned pad_amount)
{
    unsigned long long ga = (unsigned long long)(size_t)gptr;
    tdm_g0_t g0;
    g0[0] = 1u;                                            // count=1, user mode
    g0[1] = lds;                                           // lds_addr (bytes)
    g0[2] = (unsigned)(ga & 0xFFFFFFFFull);                // global_addr[31:0]
    g0[3] = (unsigned)((ga >> 32) & 0x01FFFFFFull)         // global_addr[56:32]
          | (2u << 30);                                    // type=2 ("image")
    tdm_g1_t g1;
    g1[0] = (int)((1u << 16)                               // data_size=2 bytes
          | (1u << 20)                                     // pad_enable
          | (pad_interval << 22) | (pad_amount << 25));
    g1[1] = (int)((dim0 & 0xFFFFu) << 16);                 // tensor_dim0[15:0]
    g1[2] = (int)((dim0 >> 16) | ((dim1 & 0xFFFFu) << 16));// dim0 hi | dim1 lo
    g1[3] = (int)((dim1 >> 16) | (tile0 << 16));           // dim1 hi | tile_dim0
    g1[4] = (int)tile1;                                    // tile_dim1, tile_dim2=0
    g1[5] = (int)stride0;                                  // tensor_dim0_stride lo
    g1[6] = 0;                                             // stride0 hi | stride1 lo
    g1[7] = 0;
    tdm_g2_t gz  = {0, 0, 0, 0};
    tdm_g4_t gz8 = {0, 0, 0, 0, 0, 0, 0, 0};
    __builtin_amdgcn_tensor_load_to_lds(g0, g1, gz, gz, gz8, 0);
}

// =====================================================================
// Kernel 1: fused QKV projection.  C = hs @ W_qkv + b, scattered to
// Q[BH][S][D], K[BH][S][D], V^T[BH][D][S] in bf16.
// Block: 256 thr (8 waves). Tile: 128(M) x 64(N), K-step 32.
// =====================================================================
__global__ __launch_bounds__(256) void qkv_gemm_kernel(
    const float* __restrict__ hs, const float* __restrict__ W,
    const float* __restrict__ bias,
    __bf16* __restrict__ qo, __bf16* __restrict__ ko, __bf16* __restrict__ vo)
{
    __shared__ __bf16 As[128][40];
    __shared__ __bf16 Bs[64][40];

    const int t   = threadIdx.x;
    const int wid = t >> 5, lane = t & 31, lg = lane & 15, hi = lane >> 4;
    const int n0  = blockIdx.x * 64;
    const int m0  = blockIdx.y * 128;

    v8f acc[4] = {};

    for (int k0 = 0; k0 < HID; k0 += 32) {
        { // stage A (f32 -> bf16): 128x32
            int r = t >> 1, cs = (t & 1) * 16;
            const float* src = hs + (size_t)(m0 + r) * HID + k0 + cs;
#pragma unroll
            for (int i = 0; i < 16; i += 4) {
                v4f x = *(const v4f*)(src + i);
                As[r][cs + i + 0] = f2bf(x.x); As[r][cs + i + 1] = f2bf(x.y);
                As[r][cs + i + 2] = f2bf(x.z); As[r][cs + i + 3] = f2bf(x.w);
            }
        }
        { // stage B transposed: Bs[n][k] = W[k0+k][n0+n]
            int kk = t & 31, ns = (t >> 5) * 8;
            const float* src = W + (size_t)(k0 + kk) * NQKV + n0 + ns;
#pragma unroll
            for (int i = 0; i < 8; i += 4) {
                v4f x = *(const v4f*)(src + i);
                Bs[ns + i + 0][kk] = f2bf(x.x); Bs[ns + i + 1][kk] = f2bf(x.y);
                Bs[ns + i + 2][kk] = f2bf(x.z); Bs[ns + i + 3][kk] = f2bf(x.w);
            }
        }
        __syncthreads();

        int ar = wid * 16 + lg;
        v16bf a = mk16(*(const v8bf*)&As[ar][hi * 8],
                       *(const v8bf*)&As[ar][16 + hi * 8]);
#pragma unroll
        for (int tn = 0; tn < 4; tn++) {
            int bc = tn * 16 + lg;
            v16bf bb = mk16(*(const v8bf*)&Bs[bc][hi * 8],
                            *(const v8bf*)&Bs[bc][16 + hi * 8]);
            acc[tn] = wmma_bf16(a, bb, acc[tn]);
        }
        __syncthreads();
    }

    // epilogue: bias + scatter to per-head Q/K/V layouts
#pragma unroll
    for (int tn = 0; tn < 4; tn++) {
        int col = n0 + tn * 16 + lg;
        float bq = bias[col];
        int hh = col / 384, rem = col % 384;
        int typ = rem >> 7, dd = rem & 127;
#pragma unroll
        for (int j = 0; j < 8; j++) {
            int row = m0 + wid * 16 + hi * 8 + j;
            int bb_ = row >> 11, ss = row & 2047;
            int bh  = bb_ * NHEAD + hh;
            __bf16 val = f2bf(acc[tn][j] + bq);
            if (typ == 0)      qo[((size_t)bh * SEQ + ss) * HDIM + dd] = val;
            else if (typ == 1) ko[((size_t)bh * SEQ + ss) * HDIM + dd] = val;
            else               vo[((size_t)bh * HDIM + dd) * SEQ + ss] = val;
        }
    }
}

// =====================================================================
// Kernel 2: flash attention per (b,h). Block = 128 q-rows (8 waves x 16),
// streaming over keys in 32-wide chunks.  K / V^T chunks are DMA'd into
// LDS by the Tensor Data Mover (wave 0 issues, TENSORcnt-waited, then
// workgroup barrier publishes the tile to all waves).
// =====================================================================
__global__ __launch_bounds__(256) void attn_kernel(
    const __bf16* __restrict__ qx, const __bf16* __restrict__ kx,
    const __bf16* __restrict__ vx, const float* __restrict__ alibi,
    const float* __restrict__ mask, __bf16* __restrict__ ctx)
{
    __shared__ __bf16 Ks[32][136];     // [key][d]   128 elem rows + 8 pad
    __shared__ __bf16 Vts[128][40];    // [d][key]    32 elem rows + 8 pad
    __shared__ __bf16 Pl[8][16][40];   // per-wave P transpose scratch

    const int t   = threadIdx.x;
    const int wid = t >> 5, lane = t & 31, lg = lane & 15, hi = lane >> 4;
    const int bh  = blockIdx.y;
    const int b   = bh >> 4, hh = bh & 15;
    const int q0  = blockIdx.x * 128 + wid * 16;

    // Q fragments for this wave's 16 rows (resident for whole kernel)
    v16bf aq[4];
    {
        const __bf16* qp = qx + ((size_t)bh * SEQ + q0 + lg) * HDIM;
#pragma unroll
        for (int dd = 0; dd < 4; dd++) {
            int d0 = dd * 32;
            aq[dd] = mk16(*(const v8bf*)(qp + d0 + hi * 8),
                          *(const v8bf*)(qp + d0 + 16 + hi * 8));
        }
    }

    float mrow[8], lrow[8], alpha[8];
    v8f acc[8] = {};
#pragma unroll
    for (int j = 0; j < 8; j++) { mrow[j] = -1e30f; lrow[j] = 0.f; }

    const float scale = 0.08838834764831845f;   // 2/NORM_FACTOR = 1/sqrt(128)

    const unsigned ldsK = (unsigned)(size_t)&Ks[0][0];
    const unsigned ldsV = (unsigned)(size_t)&Vts[0][0];

    for (int kn = 0; kn < SEQ; kn += 32) {
        if (wid == 0) {
            // K chunk: 32 rows x 128 bf16, row = 64 DW, LDS pad 4 DW/row
            tdm_load_2d(ldsK, kx + ((size_t)bh * SEQ + kn) * HDIM,
                        /*dim0*/HDIM, /*dim1*/32, /*tile0*/HDIM, /*tile1*/32,
                        /*stride0*/HDIM, /*pad_interval*/5, /*pad_amount*/3);
            // V^T chunk: 128 rows x 32 bf16, row = 16 DW, LDS pad 4 DW/row
            tdm_load_2d(ldsV, vx + (size_t)bh * HDIM * SEQ + kn,
                        /*dim0*/32, /*dim1*/HDIM, /*tile0*/32, /*tile1*/HDIM,
                        /*stride0*/SEQ, /*pad_interval*/3, /*pad_amount*/3);
            __builtin_amdgcn_s_wait_tensorcnt(0);
        }
        __syncthreads();

        // scores: two 16x16 tiles (keys kn..+15, kn+16..+31)
        float sv[2][8];
#pragma unroll
        for (int tile = 0; tile < 2; tile++) {
            v8f s = {};
#pragma unroll
            for (int dd = 0; dd < 4; dd++) {
                int d0 = dd * 32;
                int kc = tile * 16 + lg;
                v16bf bk = mk16(*(const v8bf*)&Ks[kc][d0 + hi * 8],
                                *(const v8bf*)&Ks[kc][d0 + 16 + hi * 8]);
                s = wmma_bf16(aq[dd], bk, s);
            }
            int colk = kn + tile * 16 + lg;
            float al = alibi[(size_t)bh * SEQ + colk];
#pragma unroll
            for (int j = 0; j < 8; j++) {
                int row = q0 + hi * 8 + j;
                float mk_ = mask[((size_t)b * SEQ + row) * SEQ + colk];
                sv[tile][j] = s[j] * scale + al + mk_;
            }
        }

        // online softmax (row reductions across the 16-lane half-groups)
#pragma unroll
        for (int j = 0; j < 8; j++) {
            float tmax = fmaxf(sv[0][j], sv[1][j]);
#pragma unroll
            for (int m_ = 1; m_ < 16; m_ <<= 1)
                tmax = fmaxf(tmax, __shfl_xor(tmax, m_, 32));
            float mn = fmaxf(mrow[j], tmax);
            alpha[j] = __expf(mrow[j] - mn);
            mrow[j]  = mn;
            float p0 = __expf(sv[0][j] - mn);
            float p1 = __expf(sv[1][j] - mn);
            sv[0][j] = p0; sv[1][j] = p1;
            float ps = p0 + p1;
#pragma unroll
            for (int m_ = 1; m_ < 16; m_ <<= 1)
                ps += __shfl_xor(ps, m_, 32);
            lrow[j] = lrow[j] * alpha[j] + ps;
        }

        // P -> bf16, transpose through per-wave LDS into A-fragment layout
#pragma unroll
        for (int tile = 0; tile < 2; tile++)
#pragma unroll
            for (int j = 0; j < 8; j++)
                Pl[wid][hi * 8 + j][tile * 16 + lg] = f2bf(sv[tile][j]);
        v16bf ap = mk16(*(const v8bf*)&Pl[wid][lg][hi * 8],
                        *(const v8bf*)&Pl[wid][lg][16 + hi * 8]);

        // rescale + P @ V  (8 d-tiles of 16)
#pragma unroll
        for (int dt = 0; dt < 8; dt++) {
#pragma unroll
            for (int j = 0; j < 8; j++) acc[dt][j] *= alpha[j];
            int dc = dt * 16 + lg;
            v16bf bv = mk16(*(const v8bf*)&Vts[dc][hi * 8],
                            *(const v8bf*)&Vts[dc][16 + hi * 8]);
            acc[dt] = wmma_bf16(ap, bv, acc[dt]);
        }
        __syncthreads();
    }

    // normalize and write context [B,S,HID] (bf16, token-major -> dense A)
#pragma unroll
    for (int dt = 0; dt < 8; dt++) {
        int dcol = hh * HDIM + dt * 16 + lg;
#pragma unroll
        for (int j = 0; j < 8; j++) {
            int row = q0 + hi * 8 + j;
            ctx[((size_t)b * SEQ + row) * HID + dcol] = f2bf(acc[dt][j] / lrow[j]);
        }
    }
}

// =====================================================================
// Kernel 3: dense projection.  out = ctx(bf16) @ W_dense + b_dense (f32 out)
// =====================================================================
__global__ __launch_bounds__(256) void dense_gemm_kernel(
    const __bf16* __restrict__ A, const float* __restrict__ W,
    const float* __restrict__ bias, float* __restrict__ out)
{
    __shared__ __bf16 As[128][40];
    __shared__ __bf16 Bs[64][40];

    const int t   = threadIdx.x;
    const int wid = t >> 5, lane = t & 31, lg = lane & 15, hi = lane >> 4;
    const int n0  = blockIdx.x * 64;
    const int m0  = blockIdx.y * 128;

    v8f acc[4] = {};

    for (int k0 = 0; k0 < HID; k0 += 32) {
        { // stage A (already bf16): 128x32
            int r = t >> 1, cs = (t & 1) * 16;
            const __bf16* src = A + (size_t)(m0 + r) * HID + k0 + cs;
            *(v8bf*)&As[r][cs]     = *(const v8bf*)(src);
            *(v8bf*)&As[r][cs + 8] = *(const v8bf*)(src + 8);
        }
        { // stage B transposed (f32 -> bf16)
            int kk = t & 31, ns = (t >> 5) * 8;
            const float* src = W + (size_t)(k0 + kk) * HID + n0 + ns;
#pragma unroll
            for (int i = 0; i < 8; i += 4) {
                v4f x = *(const v4f*)(src + i);
                Bs[ns + i + 0][kk] = f2bf(x.x); Bs[ns + i + 1][kk] = f2bf(x.y);
                Bs[ns + i + 2][kk] = f2bf(x.z); Bs[ns + i + 3][kk] = f2bf(x.w);
            }
        }
        __syncthreads();

        int ar = wid * 16 + lg;
        v16bf a = mk16(*(const v8bf*)&As[ar][hi * 8],
                       *(const v8bf*)&As[ar][16 + hi * 8]);
#pragma unroll
        for (int tn = 0; tn < 4; tn++) {
            int bc = tn * 16 + lg;
            v16bf bb = mk16(*(const v8bf*)&Bs[bc][hi * 8],
                            *(const v8bf*)&Bs[bc][16 + hi * 8]);
            acc[tn] = wmma_bf16(a, bb, acc[tn]);
        }
        __syncthreads();
    }

#pragma unroll
    for (int tn = 0; tn < 4; tn++) {
        int col = n0 + tn * 16 + lg;
        float bd = bias[col];
#pragma unroll
        for (int j = 0; j < 8; j++) {
            int row = m0 + wid * 16 + hi * 8 + j;
            out[(size_t)row * HID + col] = acc[tn][j] + bd;
        }
    }
}

// =====================================================================
extern "C" void kernel_launch(void* const* d_in, const int* in_sizes, int n_in,
                              void* d_out, int out_size, void* d_ws, size_t ws_size,
                              hipStream_t stream) {
    (void)in_sizes; (void)n_in; (void)out_size; (void)ws_size;
    const float* hs    = (const float*)d_in[0];
    const float* mask  = (const float*)d_in[1];
    const float* alibi = (const float*)d_in[2];
    const float* Wqkv  = (const float*)d_in[3];
    const float* bqkv  = (const float*)d_in[4];
    const float* Wd    = (const float*)d_in[5];
    const float* bd    = (const float*)d_in[6];
    float* out = (float*)d_out;

    const size_t QKE = (size_t)2 * NHEAD * SEQ * HDIM;  // 8,388,608 elems
    __bf16* q   = (__bf16*)d_ws;
    __bf16* k   = q + QKE;
    __bf16* v   = k + QKE;
    __bf16* ctx = v + QKE;                              // 64 MB total ws

    qkv_gemm_kernel<<<dim3(NQKV / 64, NROWS / 128), 256, 0, stream>>>(
        hs, Wqkv, bqkv, q, k, v);
    attn_kernel<<<dim3(SEQ / 128, 2 * NHEAD), 256, 0, stream>>>(
        q, k, v, alibi, mask, ctx);
    dense_gemm_kernel<<<dim3(HID / 64, NROWS / 128), 256, 0, stream>>>(
        ctx, Wd, bd, out);
}
